// Sinkhorn_38302518346113
// MI455X (gfx1250) — matrix operand order repaired and checked by reading, not compiled, not run on hardware
//
#include <hip/hip_runtime.h>

// Sinkhorn-Knopp, N=8192, 10 alternating normalizations.
// Collapsed to 5x (row-GEMV u = C@(1/v)) + 5x (col-GEMV v = C^T@(1/u))
// over a bf16 copy of (costs+eps), then out = (costs+eps) * (1/u[i]) * (1/v[j]).
// Row-GEMV uses v_wmma_f32_16x16x32_bf16: A = w broadcast to all 16 rows,
// B = 16x32 matrix tile (lane n holds matrix row n), so D[*][n] = rowdot(n).

#define KN 8192
#define EPS 1e-4f

typedef __attribute__((ext_vector_type(16))) __bf16        v16bf;
typedef __attribute__((ext_vector_type(8)))  __bf16        v8bf;
typedef __attribute__((ext_vector_type(8)))  float         v8f;
typedef __attribute__((ext_vector_type(4)))  float         f32x4;
typedef __attribute__((ext_vector_type(4)))  unsigned int  u32x4;

__device__ __forceinline__ unsigned f2bf(float f) {  // round-to-nearest-even
  unsigned u = __builtin_bit_cast(unsigned, f);
  return (u + 0x7fffu + ((u >> 16) & 1u)) >> 16;
}
__device__ __forceinline__ unsigned pack2(float a, float b) {
  return f2bf(a) | (f2bf(b) << 16);
}

// ---- costs(f32) + eps -> bf16 matrix ----
__global__ void __launch_bounds__(256) k_convert(const float* __restrict__ c,
                                                 unsigned short* __restrict__ mb) {
  const size_t i = ((size_t)blockIdx.x * 256 + threadIdx.x) * 8;
  f32x4 a = *(const f32x4*)(c + i);
  f32x4 b = *(const f32x4*)(c + i + 4);
  u32x4 o;
  o[0] = pack2(a[0] + EPS, a[1] + EPS);
  o[1] = pack2(a[2] + EPS, a[3] + EPS);
  o[2] = pack2(b[0] + EPS, b[1] + EPS);
  o[3] = pack2(b[2] + EPS, b[3] + EPS);
  *(u32x4*)(mb + i) = o;
}

// ---- w = 1.0 (bf16) ----
__global__ void __launch_bounds__(256) k_init_w(unsigned short* __restrict__ wb) {
  wb[blockIdx.x * 256 + threadIdx.x] = 0x3F80u;  // bf16 1.0
}

// ---- u[i] = sum_j mb[i,j] * wb[j]  via WMMA ----
// block = 256 thr (8 waves) = 16 rows; wave w handles cols [w*1024, w*1024+1024)
__global__ void __launch_bounds__(256) k_rowpass(const unsigned short* __restrict__ mb,
                                                 const unsigned short* __restrict__ wb,
                                                 float* __restrict__ u) {
  const int lane    = threadIdx.x & 31;
  const int wave    = threadIdx.x >> 5;
  const int rowBase = blockIdx.x << 4;
  const int colBase = wave << 10;
  const int r       = rowBase + (lane & 15);
  // B operand: lane n (and n+16) holds 16B K-slices (table order) of matrix row n
  const unsigned short* bp = mb + (size_t)r * KN + colBase + ((lane & 16) ? 8 : 0);
  // A operand: all 16 A-rows identical = w chunk; per-lane K-slice offsets
  const unsigned short* ap = wb + colBase + ((lane & 16) ? 8 : 0);
  v8f acc = {0.f, 0.f, 0.f, 0.f, 0.f, 0.f, 0.f, 0.f};
#pragma unroll 4
  for (int kc = 0; kc < 1024; kc += 32) {
    v8bf a0 = *(const v8bf*)(ap + kc);
    v8bf a1 = *(const v8bf*)(ap + kc + 16);
    v8bf b0 = *(const v8bf*)(bp + kc);
    v8bf b1 = *(const v8bf*)(bp + kc + 16);
    __builtin_prefetch(bp + kc + 512, 0, 0);
    v16bf A = __builtin_shufflevector(a0, a1, 0,1,2,3,4,5,6,7,8,9,10,11,12,13,14,15);
    v16bf B = __builtin_shufflevector(b0, b1, 0,1,2,3,4,5,6,7,8,9,10,11,12,13,14,15);
    acc = __builtin_amdgcn_wmma_f32_16x16x32_bf16(false, A, false, B, (short)0, acc,
                                                  false, false);
  }
  // D[m][n] identical over m -> lane n, acc[0] = partial rowdot(row n)
  __shared__ float red[8][16];
  if (lane < 16) red[wave][lane] = acc[0];
  __syncthreads();
  if (threadIdx.x < 16) {
    float s = 0.f;
#pragma unroll
    for (int w = 0; w < 8; ++w) s += red[w][threadIdx.x];
    u[rowBase + threadIdx.x] = s;
  }
}

// ---- rfu = 1/u ----
__global__ void __launch_bounds__(256) k_recip_u(const float* __restrict__ u,
                                                 float* __restrict__ rfu) {
  const int i = blockIdx.x * 256 + threadIdx.x;
  rfu[i] = 1.0f / u[i];
}

// ---- partial[seg][j] = sum over 64 rows of mb[i,j] * rfu[i] ----
// grid (4, 128); thread owns 8 consecutive cols (one b128 per row)
__global__ void __launch_bounds__(256) k_colpass(const unsigned short* __restrict__ mb,
                                                 const float* __restrict__ rfu,
                                                 float* __restrict__ partial) {
  const int col  = (blockIdx.x * 256 + threadIdx.x) * 8;
  const int row0 = blockIdx.y * 64;
  const unsigned short* p = mb + (size_t)row0 * KN + col;
  float acc[8] = {0.f, 0.f, 0.f, 0.f, 0.f, 0.f, 0.f, 0.f};
#pragma unroll 4
  for (int r = 0; r < 64; ++r) {
    const float s = rfu[row0 + r];
    u32x4 q = *(const u32x4*)p;
#pragma unroll
    for (int k = 0; k < 4; ++k) {
      float flo = __builtin_bit_cast(float, q[k] << 16);
      float fhi = __builtin_bit_cast(float, q[k] & 0xffff0000u);
      acc[2 * k]     = __builtin_fmaf(s, flo, acc[2 * k]);
      acc[2 * k + 1] = __builtin_fmaf(s, fhi, acc[2 * k + 1]);
    }
    p += KN;
    if ((r & 7) == 0) __builtin_prefetch(p + (size_t)KN * 8, 0, 0);
  }
  float* dst = partial + (size_t)blockIdx.y * KN + col;
  f32x4 o0 = {acc[0], acc[1], acc[2], acc[3]};
  f32x4 o1 = {acc[4], acc[5], acc[6], acc[7]};
  *(f32x4*)dst       = o0;
  *(f32x4*)(dst + 4) = o1;
}

// ---- v[j] = sum over 128 segments (deterministic tree-free fixed order) ----
__global__ void __launch_bounds__(256) k_colreduce(const float* __restrict__ partial,
                                                   float* __restrict__ v) {
  const int col = blockIdx.x * 256 + threadIdx.x;
  float s = 0.f;
#pragma unroll 8
  for (int g = 0; g < 128; ++g) s += partial[(size_t)g * KN + col];
  v[col] = s;
}

// ---- rfv = 1/v (f32) and wb = bf16(1/v) for next row pass ----
__global__ void __launch_bounds__(256) k_recip_v(const float* __restrict__ v,
                                                 unsigned short* __restrict__ wb,
                                                 float* __restrict__ rfv) {
  const int i = blockIdx.x * 256 + threadIdx.x;
  const float r = 1.0f / v[i];
  rfv[i] = r;
  wb[i]  = (unsigned short)f2bf(r);
}

// ---- out[i,j] = (costs[i,j]+eps) * rfu[i] * rfv[j] ----
__global__ void __launch_bounds__(256) k_final(const float* __restrict__ c,
                                               const float* __restrict__ rfu,
                                               const float* __restrict__ rfv,
                                               float* __restrict__ out) {
  const int col = (blockIdx.x * 256 + threadIdx.x) * 8;
  const int row = blockIdx.y;
  const float ru = rfu[row];
  const size_t off = (size_t)row * KN + col;
  f32x4 c0 = *(const f32x4*)(c + off);
  f32x4 c1 = *(const f32x4*)(c + off + 4);
  f32x4 w0 = *(const f32x4*)(rfv + col);
  f32x4 w1 = *(const f32x4*)(rfv + col + 4);
  f32x4 r0 = (c0 + EPS) * w0 * ru;
  f32x4 r1 = (c1 + EPS) * w1 * ru;
  *(f32x4*)(out + off)     = r0;
  *(f32x4*)(out + off + 4) = r1;
}

extern "C" void kernel_launch(void* const* d_in, const int* in_sizes, int n_in,
                              void* d_out, int out_size, void* d_ws, size_t ws_size,
                              hipStream_t stream) {
  const float* costs = (const float*)d_in[0];
  float* out = (float*)d_out;

  // workspace layout (~133 MB)
  char* ws = (char*)d_ws;
  unsigned short* mb = (unsigned short*)ws;                    // 128 MB bf16 matrix
  size_t off = (size_t)KN * KN * sizeof(unsigned short);
  float* u   = (float*)(ws + off);  off += (size_t)KN * 4;
  float* v   = (float*)(ws + off);  off += (size_t)KN * 4;
  float* rfu = (float*)(ws + off);  off += (size_t)KN * 4;
  float* rfv = (float*)(ws + off);  off += (size_t)KN * 4;
  unsigned short* wb = (unsigned short*)(ws + off);
  off += (size_t)KN * 2;
  off = (off + 255) & ~(size_t)255;
  float* partial = (float*)(ws + off);                         // 128*KN f32 = 4 MB

  k_convert<<<dim3((unsigned)(((size_t)KN * KN) / (256 * 8))), 256, 0, stream>>>(costs, mb);
  k_init_w<<<KN / 256, 256, 0, stream>>>(wb);

  for (int it = 0; it < 5; ++it) {
    k_rowpass<<<KN / 16, 256, 0, stream>>>(mb, wb, u);          // u = C @ (1/v)
    k_recip_u<<<KN / 256, 256, 0, stream>>>(u, rfu);
    k_colpass<<<dim3(4, 128), 256, 0, stream>>>(mb, rfu, partial);  // v = C^T @ (1/u)
    k_colreduce<<<KN / 256, 256, 0, stream>>>(partial, v);
    k_recip_v<<<KN / 256, 256, 0, stream>>>(v, wb, rfv);
  }

  k_final<<<dim3(4, KN), 256, 0, stream>>>(costs, rfu, rfv, out);
}
